// GraphConvLayer_25512105738334
// MI455X (gfx1250) — matrix operand order repaired and checked by reading, not compiled, not run on hardware
//
#include <hip/hip_runtime.h>
#include <hip/hip_bf16.h>
#include <stdint.h>

typedef __attribute__((ext_vector_type(16))) __bf16 v16bf;
typedef __attribute__((ext_vector_type(8)))  float  v8f;

#define D               128
#define WAVES_PER_BLOCK 8
#define ROWS_PER_WAVE   16
#define ROWS_PER_BLOCK  (WAVES_PER_BLOCK * ROWS_PER_WAVE)

// ---------------------------------------------------------------------------
// Kernel 0: zero the output (harness poisons d_out with 0xAA before timing)
// ---------------------------------------------------------------------------
__global__ void __launch_bounds__(256) gcn_zero(float4* __restrict__ out, int n4) {
    const int i = blockIdx.x * 256 + threadIdx.x;
    if (i < n4) out[i] = make_float4(0.f, 0.f, 0.f, 0.f);
}

// ---------------------------------------------------------------------------
// Kernel 1: h = x @ W^T + b  via v_wmma_f32_16x16x32_bf16
//   - W (128x128 f32) is converted once to bf16 in LDS (32 KB of 320 KB WGP LDS)
//   - one wave computes a 16x128 strip of h (8 accumulator tiles, 4 K-steps)
//   - accumulators are initialized with the bias (bias depends only on the
//     output column o = t*16 + (lane&15), constant across the 8 C elements)
// ---------------------------------------------------------------------------
__global__ void __launch_bounds__(256) gcn_gemm_bias(
    const float* __restrict__ x, const float* __restrict__ W,
    const float* __restrict__ bias, float* __restrict__ h, int nrows)
{
    __shared__ __align__(32) __bf16 Wlds[D * D];   // 32 KB, [o][d] row-major
    __shared__ float blds[D];

    const int tid = threadIdx.x;

    // Cooperative f32 -> bf16 conversion of W into LDS
    for (int i = tid * 4; i < D * D; i += 256 * 4) {
        float4 w4 = *(const float4*)(W + i);
        Wlds[i + 0] = (__bf16)w4.x;
        Wlds[i + 1] = (__bf16)w4.y;
        Wlds[i + 2] = (__bf16)w4.z;
        Wlds[i + 3] = (__bf16)w4.w;
    }
    if (tid < D) blds[tid] = bias[tid];
    __syncthreads();

    const int wave = tid >> 5;
    const int lane = tid & 31;
    const int row0 = blockIdx.x * ROWS_PER_BLOCK + wave * ROWS_PER_WAVE;
    if (row0 >= nrows) return;          // uniform per wave -> EXEC stays all-1s

    const int n = lane & 15;            // A row m / B,C,D column n
    const int g = lane >> 4;            // half-wave group

    // Init accumulators with bias
    v8f acc[8];
#pragma unroll
    for (int t = 0; t < 8; ++t) {
        const float bv = blds[t * 16 + n];
#pragma unroll
        for (int j = 0; j < 8; ++j) acc[t][j] = bv;
    }

    const float* xrow = x + (size_t)(row0 + n) * D;

#pragma unroll
    for (int k0 = 0; k0 < D; k0 += 32) {
        // A fragment (16x32 bf16): lanes 0-15 hold K {0..15},
        // lanes 16-31 hold K {16..31} of row m = lane&15
        const float* pa = xrow + k0 + 16 * g;
        float4 a0 = *(const float4*)(pa + 0);
        float4 a1 = *(const float4*)(pa + 4);
        float4 a2 = *(const float4*)(pa + 8);
        float4 a3 = *(const float4*)(pa + 12);
        v16bf a;
        a[0]  = (__bf16)a0.x; a[1]  = (__bf16)a0.y; a[2]  = (__bf16)a0.z; a[3]  = (__bf16)a0.w;
        a[4]  = (__bf16)a1.x; a[5]  = (__bf16)a1.y; a[6]  = (__bf16)a1.z; a[7]  = (__bf16)a1.w;
        a[8]  = (__bf16)a2.x; a[9]  = (__bf16)a2.y; a[10] = (__bf16)a2.z; a[11] = (__bf16)a2.w;
        a[12] = (__bf16)a3.x; a[13] = (__bf16)a3.y; a[14] = (__bf16)a3.z; a[15] = (__bf16)a3.w;

#pragma unroll
        for (int t = 0; t < 8; ++t) {
            // B fragment (32x16 bf16): lane holds 16 consecutive K values of
            // column n; B[k][n] = W[o = t*16+n][d = k0+k]  (row-major in LDS)
            const v16bf bf = *(const v16bf*)&Wlds[(t * 16 + n) * D + k0 + 16 * g];
            acc[t] = __builtin_amdgcn_wmma_f32_16x16x32_bf16(
                         false, a, false, bf, (short)0, acc[t], false, false);
        }
    }

    // Store C tiles: element j of acc -> (m = j + 8*g, o = t*16 + n)
#pragma unroll
    for (int t = 0; t < 8; ++t) {
        const int o = t * 16 + n;
#pragma unroll
        for (int j = 0; j < 8; ++j) {
            const int m = j + 8 * g;
            h[(size_t)(row0 + m) * D + o] = acc[t][j];
        }
    }
}

// ---------------------------------------------------------------------------
// Kernel 2: out[row[e]] += val[e] * h[col[e]]  (one wave per edge)
// Device-scope hardware f32 atomics (L2-side RMW; h and out are L2-resident)
// ---------------------------------------------------------------------------
__device__ __forceinline__ void gatomic_add_f32(float* p, float v) {
    asm volatile("global_atomic_add_f32 %0, %1, off scope:SCOPE_DEV"
                 :: "v"((uint64_t)(uintptr_t)p), "v"(v) : "memory");
}

__global__ void __launch_bounds__(256) gcn_scatter(
    const int* __restrict__ erow, const int* __restrict__ ecol,
    const float* __restrict__ eval, const float* __restrict__ h,
    float* __restrict__ out, int nedges)
{
    const int wid  = (int)((blockIdx.x * 256u + (unsigned)threadIdx.x) >> 5);
    const int lane = threadIdx.x & 31;
    if (wid >= nedges) return;

    const int   r = erow[wid];
    const int   c = ecol[wid];
    const float v = eval[wid];

    const float4 msg = *(const float4*)(h + (size_t)c * D + lane * 4);
    float* dst = out + (size_t)r * D + lane * 4;
    gatomic_add_f32(dst + 0, v * msg.x);
    gatomic_add_f32(dst + 1, v * msg.y);
    gatomic_add_f32(dst + 2, v * msg.z);
    gatomic_add_f32(dst + 3, v * msg.w);
}

// ---------------------------------------------------------------------------
extern "C" void kernel_launch(void* const* d_in, const int* in_sizes, int n_in,
                              void* d_out, int out_size, void* d_ws, size_t ws_size,
                              hipStream_t stream) {
    const float* x    = (const float*)d_in[0];
    const int*   erow = (const int*)d_in[1];
    const int*   ecol = (const int*)d_in[2];
    const float* eval = (const float*)d_in[3];
    const float* W    = (const float*)d_in[4];
    const float* bias = (const float*)d_in[5];
    float* out = (float*)d_out;
    float* h   = (float*)d_ws;                 // N*128 f32 = 51.2 MB scratch

    const int nrows  = in_sizes[0] / D;        // 100000
    const int nedges = in_sizes[1];            // 3200000

    const int n4 = (nrows * D) / 4;
    gcn_zero<<<(n4 + 255) / 256, 256, 0, stream>>>((float4*)out, n4);

    const int gemm_blocks = (nrows + ROWS_PER_BLOCK - 1) / ROWS_PER_BLOCK;
    gcn_gemm_bias<<<gemm_blocks, 256, 0, stream>>>(x, W, bias, h, nrows);

    const long long sc_threads = (long long)nedges * 32;
    gcn_scatter<<<(int)((sc_threads + 255) / 256), 256, 0, stream>>>(
        erow, ecol, eval, h, out, nedges);
}
